// Predictor_51969104281860
// MI455X (gfx1250) — matrix operand order repaired and checked by reading, not compiled, not run on hardware
//
#include <hip/hip_runtime.h>
#include <hip/hip_bf16.h>

// ---------- types ----------
typedef __attribute__((ext_vector_type(16))) __bf16 v16bf;
typedef __attribute__((ext_vector_type(8)))  float  v8f;

union Frag {
    v16bf v;
    uint4 q[2];
};

// ---------- bf16 helpers (round-to-nearest-even) ----------
static __device__ __forceinline__ unsigned short f2bf(float f) {
    union { float f; unsigned u; } x; x.f = f;
    unsigned r = x.u + 0x7FFFu + ((x.u >> 16) & 1u);
    return (unsigned short)(r >> 16);
}
static __device__ __forceinline__ float bf2f(unsigned short s) {
    union { unsigned u; float f; } x; x.u = ((unsigned)s) << 16;
    return x.f;
}
static __device__ __forceinline__ float sigmoidf_fast(float x) {
    return 1.0f / (1.0f + __expf(-x));        // saturates cleanly at 0/1
}
static __device__ __forceinline__ float tanhf_fast(float x) {
    return 1.0f - 2.0f / (__expf(2.0f * x) + 1.0f);  // saturates cleanly at +/-1
}

// ---------- f32 -> bf16 conversion ----------
__global__ __launch_bounds__(256) void convert_f32_bf16(
    const float* __restrict__ src, unsigned short* __restrict__ dst, int n) {
    for (int i = blockIdx.x * 256 + threadIdx.x; i < n; i += gridDim.x * 256)
        dst[i] = f2bf(src[i]);
}

// ---------- generic bf16 WMMA GEMM: C[M,N] = act(A[M,K] @ W[N,K]^T + bias (+bias2)) ----------
// grid: (M/16, N/64), block: 128 threads (4 waves), each wave owns one 16x16 N-tile.
__global__ __launch_bounds__(128) void gemm_bf16_wmma(
    const unsigned short* __restrict__ A,   // [M,K] bf16 row-major
    const unsigned short* __restrict__ W,   // [N,K] bf16 row-major (B = W^T)
    const float* __restrict__ bias,         // [N]
    const float* __restrict__ bias2,        // [N] or nullptr
    unsigned short* __restrict__ outBf,     // [M,N] bf16 or nullptr
    float* __restrict__ outF,               // [M,N] f32  or nullptr
    int N, int K, int relu)
{
    const int lane = threadIdx.x & 31;
    const int w    = threadIdx.x >> 5;
    const int m0   = blockIdx.x * 16;
    const int nb   = blockIdx.y * 64 + w * 16;
    const int mlo  = lane & 15;   // row (A) / column (B) within tile
    const int kh   = lane >> 4;   // K half-select: lane half kh holds k = kh*16..kh*16+15 (this code's order)

    v8f acc = {0.f, 0.f, 0.f, 0.f, 0.f, 0.f, 0.f, 0.f};

    const int nkc = K >> 5;
    for (int kc = 0; kc < nkc; ++kc) {
        Frag a, b;
        // A 16x32 bf16 fragment: lane m%16 holds row m; 16 contiguous K at kh*16
        const unsigned short* ap = A + (size_t)(m0 + mlo) * K + kc * 32 + kh * 16;
        a.q[0] = *(const uint4*)(ap);
        a.q[1] = *(const uint4*)(ap + 8);
        // B 32x16 bf16 fragment: lane holds column n, 16 contiguous K at kh*16
        const unsigned short* bp = W + (size_t)(nb + mlo) * K + kc * 32 + kh * 16;
        b.q[0] = *(const uint4*)(bp);
        b.q[1] = *(const uint4*)(bp + 8);
        __builtin_prefetch(bp + 32, 0, 1);
        acc = __builtin_amdgcn_wmma_f32_16x16x32_bf16(
            false, a.v, false, b.v, (short)0, acc, false, false);
    }

    const int n = nb + mlo;
    float bsum = bias[n] + (bias2 ? bias2[n] : 0.0f);
    for (int r = 0; r < 8; ++r) {
        // C layout: VGPR r -> M = r (lanes 0-15) / M = 8+r (lanes 16-31), N = lane%16
        int m = m0 + kh * 8 + r;
        float v = acc[r] + bsum;
        if (relu) v = fmaxf(v, 0.0f);
        if (outF)  outF [(size_t)m * N + n] = v;
        if (outBf) outBf[(size_t)m * N + n] = f2bf(v);
    }
}

// ---------- persistent LSTM: each block owns 16 batch rows for all 50 steps ----------
// grid: 256 blocks (4096/16), block: 256 threads (8 waves).
// Wave w computes gate columns [w*128, (w+1)*128): waves 0-1 = i, 2-3 = f, 4-5 = g, 6-7 = o.
__global__ __launch_bounds__(256) void lstm_persistent(
    const float* __restrict__ xpart,          // [4096,1024] f32: z_embed@W_ih^T + b_ih + b_hh
    const unsigned short* __restrict__ Whh,   // [1024,256] bf16
    float* __restrict__ out)                  // [4096,50,256] f32
{
    __shared__ unsigned short h_lds[16 * 256];   // 8 KB: h_t tile, bf16
    __shared__ unsigned short g_lds[16 * 1024];  // 32 KB: activated gates, bf16

    const int tid  = threadIdx.x;
    const int lane = tid & 31;
    const int w    = tid >> 5;
    const int mlo  = lane & 15;
    const int kh   = lane >> 4;
    const int r0   = blockIdx.x * 16;

    // h0 = 0
    for (int i = tid; i < 16 * 256; i += 256) h_lds[i] = 0;

    // Hoist x_part gather out of the time loop: identical every step (input is constant).
    // xp[nt][r] lives in registers (64 VGPRs/lane).
    float xp[8][8];
    #pragma unroll
    for (int nt = 0; nt < 8; ++nt) {
        const int gb = (w * 8 + nt) * 16;
        #pragma unroll
        for (int r = 0; r < 8; ++r) {
            int m = kh * 8 + r;
            xp[nt][r] = xpart[(size_t)(r0 + m) * 1024 + gb + mlo];
        }
    }

    // elementwise-phase mapping: thread -> (row, 16-wide contiguous j block); c in registers
    const int row = tid & 15;
    const int jb  = (tid >> 4) * 16;
    float c[16];
    #pragma unroll
    for (int k = 0; k < 16; ++k) c[k] = 0.0f;

    const bool gate_is_tanh = (w == 4) || (w == 5);  // g-gate waves

    for (int step = 0; step < 50; ++step) {
        __syncthreads();  // h_lds ready for this step's GEMM

        // Hoist A fragments (h tile) out of the nt loop: loaded once, reused by 8 tiles.
        Frag a[8];
        #pragma unroll
        for (int kc = 0; kc < 8; ++kc) {
            const unsigned short* ap = &h_lds[mlo * 256 + kc * 32 + kh * 16];
            a[kc].q[0] = *(const uint4*)(ap);
            a[kc].q[1] = *(const uint4*)(ap + 8);
        }

        // gates = h @ Whh^T + xpart, activated -> g_lds
        #pragma unroll
        for (int nt = 0; nt < 8; ++nt) {
            const int gb = (w * 8 + nt) * 16;  // global gate-column base
            v8f acc = {0.f, 0.f, 0.f, 0.f, 0.f, 0.f, 0.f, 0.f};
            #pragma unroll
            for (int kc = 0; kc < 8; ++kc) {
                Frag b;
                const unsigned short* bp = Whh + (size_t)(gb + mlo) * 256 + kc * 32 + kh * 16;
                b.q[0] = *(const uint4*)(bp);
                b.q[1] = *(const uint4*)(bp + 8);
                __builtin_prefetch(bp + 32, 0, 1);
                acc = __builtin_amdgcn_wmma_f32_16x16x32_bf16(
                    false, a[kc].v, false, b.v, (short)0, acc, false, false);
            }
            #pragma unroll
            for (int r = 0; r < 8; ++r) {
                int m = kh * 8 + r;
                float g = acc[r] + xp[nt][r];
                float av = gate_is_tanh ? tanhf_fast(g) : sigmoidf_fast(g);
                g_lds[m * 1024 + gb + mlo] = f2bf(av);
            }
        }
        __syncthreads();  // gates ready

        // c/h update + vectorized output write + refresh h_lds for next step
        union { float4 v4[4]; float f[16]; } hv;
        #pragma unroll
        for (int k = 0; k < 16; ++k) {
            int j = jb + k;
            float ig = bf2f(g_lds[row * 1024 +       j]);
            float fg = bf2f(g_lds[row * 1024 + 256 + j]);
            float gg = bf2f(g_lds[row * 1024 + 512 + j]);
            float og = bf2f(g_lds[row * 1024 + 768 + j]);
            float cv = fg * c[k] + ig * gg;
            c[k] = cv;
            float h = og * tanhf_fast(cv);
            hv.f[k] = h;
            h_lds[row * 256 + j] = f2bf(h);
        }
        float* orow = out + (size_t)(r0 + row) * (50 * 256) + (size_t)step * 256 + jb;
        #pragma unroll
        for (int q = 0; q < 4; ++q)
            *(float4*)(orow + q * 4) = hv.v4[q];
    }
}

// ---------- host-side orchestration ----------
extern "C" void kernel_launch(void* const* d_in, const int* in_sizes, int n_in,
                              void* d_out, int out_size, void* d_ws, size_t ws_size,
                              hipStream_t stream) {
    (void)in_sizes; (void)n_in; (void)out_size; (void)ws_size;

    const float* zs   = (const float*)d_in[0];   // [4096,256]
    const float* W1   = (const float*)d_in[1];   // [512,256]
    const float* b1   = (const float*)d_in[2];
    const float* W2   = (const float*)d_in[3];   // [256,512]
    const float* b2   = (const float*)d_in[4];
    const float* W3   = (const float*)d_in[5];   // [128,256]
    const float* b3   = (const float*)d_in[6];
    const float* W_ih = (const float*)d_in[7];   // [1024,128]
    const float* W_hh = (const float*)d_in[8];   // [1024,256]
    const float* b_ih = (const float*)d_in[9];
    const float* b_hh = (const float*)d_in[10];
    float* out = (float*)d_out;

    // workspace carve-up (256B aligned)
    char* ws = (char*)d_ws;
    size_t off = 0;
    auto carve = [&](size_t bytes) -> char* {
        char* p = ws + off;
        off += (bytes + 255) & ~(size_t)255;
        return p;
    };
    unsigned short* zs_bf  = (unsigned short*)carve((size_t)4096 * 256 * 2);
    unsigned short* W1_bf  = (unsigned short*)carve((size_t)512 * 256 * 2);
    unsigned short* W2_bf  = (unsigned short*)carve((size_t)256 * 512 * 2);
    unsigned short* W3_bf  = (unsigned short*)carve((size_t)128 * 256 * 2);
    unsigned short* Wih_bf = (unsigned short*)carve((size_t)1024 * 128 * 2);
    unsigned short* Whh_bf = (unsigned short*)carve((size_t)1024 * 256 * 2);
    unsigned short* h1_bf  = (unsigned short*)carve((size_t)4096 * 512 * 2);
    unsigned short* h2_bf  = (unsigned short*)carve((size_t)4096 * 256 * 2);
    unsigned short* ze_bf  = (unsigned short*)carve((size_t)4096 * 128 * 2);
    float*          xpart  = (float*)carve((size_t)4096 * 1024 * 4);

    auto conv = [&](const float* s, unsigned short* d, int n) {
        convert_f32_bf16<<<(n + 255) / 256, 256, 0, stream>>>(s, d, n);
    };
    conv(zs,   zs_bf,  4096 * 256);
    conv(W1,   W1_bf,  512 * 256);
    conv(W2,   W2_bf,  256 * 512);
    conv(W3,   W3_bf,  128 * 256);
    conv(W_ih, Wih_bf, 1024 * 128);
    conv(W_hh, Whh_bf, 1024 * 256);

    // h1 = relu(zs @ W1^T + b1)            [4096,512]
    gemm_bf16_wmma<<<dim3(4096 / 16, 512 / 64), 128, 0, stream>>>(
        zs_bf, W1_bf, b1, nullptr, h1_bf, nullptr, 512, 256, 1);
    // h2 = relu(h1 @ W2^T + b2)            [4096,256]
    gemm_bf16_wmma<<<dim3(4096 / 16, 256 / 64), 128, 0, stream>>>(
        h1_bf, W2_bf, b2, nullptr, h2_bf, nullptr, 256, 512, 1);
    // z_embed = h2 @ W3^T + b3             [4096,128]
    gemm_bf16_wmma<<<dim3(4096 / 16, 128 / 64), 128, 0, stream>>>(
        h2_bf, W3_bf, b3, nullptr, ze_bf, nullptr, 128, 256, 0);
    // x_part = z_embed @ W_ih^T + b_ih + b_hh   [4096,1024] f32
    gemm_bf16_wmma<<<dim3(4096 / 16, 1024 / 64), 128, 0, stream>>>(
        ze_bf, Wih_bf, b_ih, b_hh, nullptr, xpart, 1024, 128, 0);

    // persistent 50-step LSTM, 16 batch rows per block
    lstm_persistent<<<4096 / 16, 256, 0, stream>>>(xpart, Whh_bf, out);
}
